// RoutedSharedMoEFFN_62964220559756
// MI455X (gfx1250) — matrix operand order repaired and checked by reading, not compiled, not run on hardware
//
#include <hip/hip_runtime.h>
#include <hip/hip_bf16.h>
#include <stdint.h>

typedef __bf16 bf16;
typedef __attribute__((ext_vector_type(16))) __bf16 v16bf;
typedef __attribute__((ext_vector_type(8)))  float  v8f;

#define D_DIM  1024
#define E_NUM  32
#define H_DIM  128
#define HS_DIM 512
#define K_TOP  4
#define M_TILE 32

union ABFrag { v16bf v; uint4 q[2]; };

__device__ __forceinline__ v8f wmma_bf16(const ABFrag& a, const ABFrag& b, v8f c) {
  return __builtin_amdgcn_wmma_f32_16x16x32_bf16(false, a.v, false, b.v, (short)0, c, false, false);
}

// ---------------- fp32 -> bf16 conversion (vectorized x4) ----------------
__global__ void cvt_bf16_kernel(const float* __restrict__ src, bf16* __restrict__ dst, int n4) {
  int i = blockIdx.x * blockDim.x + threadIdx.x;
  if (i >= n4) return;
  float4 f = ((const float4*)src)[i];
  union { bf16 h[4]; ushort4 u; } cv;
  cv.h[0] = (bf16)f.x; cv.h[1] = (bf16)f.y; cv.h[2] = (bf16)f.z; cv.h[3] = (bf16)f.w;
  ((ushort4*)dst)[i] = cv.u;
}

__global__ void zero_i32_kernel(int* __restrict__ p, int n) {
  int i = blockIdx.x * blockDim.x + threadIdx.x;
  if (i < n) p[i] = 0;
}

// ---------------- router: logits, softmax, top-4, renorm -----------------
__global__ __launch_bounds__(256) void router_kernel(
    const float* __restrict__ x, const float* __restrict__ rw, const float* __restrict__ rb,
    int* __restrict__ tki, float* __restrict__ tks, int* __restrict__ counts) {
  __shared__ float red[256];
  __shared__ float logit[E_NUM];
  __shared__ float prob[E_NUM];
  int n = blockIdx.x;
  int tid = threadIdx.x;
  int e = tid >> 3, j = tid & 7;
  const float* xr = x + (size_t)n * D_DIM;
  const float* wr = rw + (size_t)e * D_DIM;
  float p = 0.f;
  for (int d = j * 4; d < D_DIM; d += 32) {
    float4 xv = *(const float4*)(xr + d);
    float4 wv = *(const float4*)(wr + d);
    p += xv.x * wv.x + xv.y * wv.y + xv.z * wv.z + xv.w * wv.w;
  }
  red[tid] = p;
  __syncthreads();
  if (j == 0) {
    float s = 0.f;
    for (int q = 0; q < 8; ++q) s += red[e * 8 + q];
    logit[e] = s + rb[e];
  }
  __syncthreads();
  if (tid == 0) {
    float mx = logit[0];
    for (int q = 1; q < E_NUM; ++q) mx = fmaxf(mx, logit[q]);
    float se = 0.f;
    for (int q = 0; q < E_NUM; ++q) { prob[q] = __expf(logit[q] - mx); se += prob[q]; }
    float inv = 1.f / se;
    int   idx[K_TOP]; float val[K_TOP]; float s4 = 0.f;
    for (int k = 0; k < K_TOP; ++k) {
      int am = 0; float mv = -1.f;
      for (int q = 0; q < E_NUM; ++q) { float v = prob[q]; if (v > mv) { mv = v; am = q; } }
      prob[am] = -2.f;
      idx[k] = am; val[k] = mv * inv; s4 += mv * inv;
    }
    float rinv = 1.f / fmaxf(s4, 1e-12f);
    for (int k = 0; k < K_TOP; ++k) {
      tki[n * K_TOP + k] = idx[k];
      tks[n * K_TOP + k] = val[k] * rinv;
      atomicAdd(&counts[idx[k]], 1);
    }
  }
}

__global__ void offsets_kernel(const int* __restrict__ counts, int* __restrict__ offs,
                               int* __restrict__ cursor) {
  if (threadIdx.x == 0) {
    int s = 0;
    for (int e = 0; e < E_NUM; ++e) { offs[e] = s; s += counts[e]; }
  }
  if (threadIdx.x < E_NUM) cursor[threadIdx.x] = 0;
}

__global__ void scatter_kernel(const int* __restrict__ tki, const float* __restrict__ tks,
                               const int* __restrict__ offs, int* __restrict__ cursor,
                               int* __restrict__ ptok, float* __restrict__ psc, int total) {
  int i = blockIdx.x * blockDim.x + threadIdx.x;
  if (i >= total) return;
  int n = i >> 2;
  int e = tki[i];
  int pos = atomicAdd(&cursor[e], 1);
  ptok[offs[e] + pos] = n;
  psc[offs[e] + pos]  = tks[i];
}

// ---------------- shared expert: SwiGLU, writes (initializes) out ----------------
// M=32 token tile per block: B fragments reused across two A row-tiles.
__global__ __launch_bounds__(256) void shared_expert_kernel(
    const bf16* __restrict__ xb, const bf16* __restrict__ sg,
    const bf16* __restrict__ su, const bf16* __restrict__ sd,
    float* __restrict__ out) {
  __shared__ __align__(16) bf16 hbuf[M_TILE][HS_DIM];   // 32 KB
  int row0 = blockIdx.x * M_TILE;
  int lane = threadIdx.x & 31;
  int wave = threadIdx.x >> 5;
  int nlo = lane & 15;
  int khi = lane >> 4;
  const bf16* arow0 = xb + (size_t)(row0 + nlo) * D_DIM;
  const bf16* arow1 = xb + (size_t)(row0 + 16 + nlo) * D_DIM;

  // gate/up: 32 x 512 output, 32 n-tiles over 8 waves (4 iters each)
  for (int nt = wave; nt < HS_DIM / 16; nt += 8) {
    v8f cg0 = {}; v8f cg1 = {}; v8f cu0 = {}; v8f cu1 = {};
    const bf16* bg_base = sg + (size_t)(nt * 16 + nlo) * D_DIM + khi * 8;
    const bf16* bu_base = su + (size_t)(nt * 16 + nlo) * D_DIM + khi * 8;
    for (int kb = 0; kb < D_DIM; kb += 32) {
      ABFrag a0, a1, bg, bu;
      a0.q[0] = *(const uint4*)(arow0 + kb + khi * 8);
      a0.q[1] = *(const uint4*)(arow0 + kb + 16 + khi * 8);
      a1.q[0] = *(const uint4*)(arow1 + kb + khi * 8);
      a1.q[1] = *(const uint4*)(arow1 + kb + 16 + khi * 8);
      bg.q[0] = *(const uint4*)(bg_base + kb);
      bg.q[1] = *(const uint4*)(bg_base + kb + 16);
      bu.q[0] = *(const uint4*)(bu_base + kb);
      bu.q[1] = *(const uint4*)(bu_base + kb + 16);
      cg0 = wmma_bf16(a0, bg, cg0);
      cg1 = wmma_bf16(a1, bg, cg1);
      cu0 = wmma_bf16(a0, bu, cu0);
      cu1 = wmma_bf16(a1, bu, cu1);
    }
#pragma unroll
    for (int r = 0; r < 8; ++r) {
      float g0 = cg0[r], u0 = cu0[r];
      float g1 = cg1[r], u1 = cu1[r];
      hbuf[khi * 8 + r][nt * 16 + nlo]      = (bf16)((g0 / (1.f + __expf(-g0))) * u0);
      hbuf[16 + khi * 8 + r][nt * 16 + nlo] = (bf16)((g1 / (1.f + __expf(-g1))) * u1);
    }
  }
  __syncthreads();

  // down: 32 x 1024 output, K = 512
  for (int nt = wave; nt < D_DIM / 16; nt += 8) {
    v8f c0 = {}; v8f c1 = {};
    const bf16* b_base  = sd + (size_t)(nt * 16 + nlo) * HS_DIM + khi * 8;
    const bf16* a_base0 = &hbuf[nlo][0];
    const bf16* a_base1 = &hbuf[16 + nlo][0];
    for (int kb = 0; kb < HS_DIM; kb += 32) {
      ABFrag a0, a1, b;
      a0.q[0] = *(const uint4*)(a_base0 + kb + khi * 8);
      a0.q[1] = *(const uint4*)(a_base0 + kb + 16 + khi * 8);
      a1.q[0] = *(const uint4*)(a_base1 + kb + khi * 8);
      a1.q[1] = *(const uint4*)(a_base1 + kb + 16 + khi * 8);
      b.q[0]  = *(const uint4*)(b_base + kb);
      b.q[1]  = *(const uint4*)(b_base + kb + 16);
      c0 = wmma_bf16(a0, b, c0);
      c1 = wmma_bf16(a1, b, c1);
    }
#pragma unroll
    for (int r = 0; r < 8; ++r) {
      out[(size_t)(row0 + khi * 8 + r) * D_DIM + nt * 16 + nlo]      = c0[r];
      out[(size_t)(row0 + 16 + khi * 8 + r) * D_DIM + nt * 16 + nlo] = c1[r];
    }
  }
}

// ---------------- routed experts: gathered 32-token tiles, atomic-add into out ----------------
__global__ __launch_bounds__(256) void routed_expert_kernel(
    const bf16* __restrict__ xb, const bf16* __restrict__ gw,
    const bf16* __restrict__ uw, const bf16* __restrict__ dw,
    const int* __restrict__ ptok, const float* __restrict__ psc,
    const int* __restrict__ offs, const int* __restrict__ counts,
    float* __restrict__ out) {
  int e = blockIdx.y;
  int cnt = counts[e];
  int t0 = blockIdx.x * M_TILE;
  if (t0 >= cnt) return;
  __shared__ int   tokS[M_TILE];
  __shared__ float scS[M_TILE];
  __shared__ __align__(16) bf16 hbuf[M_TILE][H_DIM];    // 8 KB
  if (threadIdx.x < M_TILE) {
    int i = t0 + threadIdx.x;
    int off = offs[e];
    if (i < cnt) { tokS[threadIdx.x] = ptok[off + i]; scS[threadIdx.x] = psc[off + i]; }
    else         { tokS[threadIdx.x] = -1;            scS[threadIdx.x] = 0.f; }
  }
  __syncthreads();
  int lane = threadIdx.x & 31;
  int wave = threadIdx.x >> 5;
  int nlo = lane & 15, khi = lane >> 4;
  int tk0 = tokS[nlo];
  int tk1 = tokS[16 + nlo];
  const bf16* arow0 = xb + (size_t)(tk0 < 0 ? 0 : tk0) * D_DIM;
  const bf16* arow1 = xb + (size_t)(tk1 < 0 ? 0 : tk1) * D_DIM;
  const bf16* gwe = gw + (size_t)e * H_DIM * D_DIM;
  const bf16* uwe = uw + (size_t)e * H_DIM * D_DIM;
  const bf16* dwe = dw + (size_t)e * D_DIM * H_DIM;

  // gate/up: H=128 -> exactly one n-tile per wave; 32x128 h tile
  {
    int nt = wave;
    v8f cg0 = {}; v8f cg1 = {}; v8f cu0 = {}; v8f cu1 = {};
    const bf16* bg_base = gwe + (size_t)(nt * 16 + nlo) * D_DIM + khi * 8;
    const bf16* bu_base = uwe + (size_t)(nt * 16 + nlo) * D_DIM + khi * 8;
    for (int kb = 0; kb < D_DIM; kb += 32) {
      ABFrag a0, a1, bg, bu;
      a0.q[0] = *(const uint4*)(arow0 + kb + khi * 8);
      a0.q[1] = *(const uint4*)(arow0 + kb + 16 + khi * 8);
      a1.q[0] = *(const uint4*)(arow1 + kb + khi * 8);
      a1.q[1] = *(const uint4*)(arow1 + kb + 16 + khi * 8);
      bg.q[0] = *(const uint4*)(bg_base + kb);
      bg.q[1] = *(const uint4*)(bg_base + kb + 16);
      bu.q[0] = *(const uint4*)(bu_base + kb);
      bu.q[1] = *(const uint4*)(bu_base + kb + 16);
      cg0 = wmma_bf16(a0, bg, cg0);
      cg1 = wmma_bf16(a1, bg, cg1);
      cu0 = wmma_bf16(a0, bu, cu0);
      cu1 = wmma_bf16(a1, bu, cu1);
    }
#pragma unroll
    for (int r = 0; r < 8; ++r) {
      float g0 = cg0[r], u0 = cu0[r];
      float g1 = cg1[r], u1 = cu1[r];
      hbuf[khi * 8 + r][nt * 16 + nlo]      = (bf16)((g0 / (1.f + __expf(-g0))) * u0);
      hbuf[16 + khi * 8 + r][nt * 16 + nlo] = (bf16)((g1 / (1.f + __expf(-g1))) * u1);
    }
  }
  __syncthreads();

  // down: 32 x 1024 output, K = 128 (4 WMMA k-steps), scaled atomic add
  for (int nt = wave; nt < D_DIM / 16; nt += 8) {
    v8f c0 = {}; v8f c1 = {};
    const bf16* b_base  = dwe + (size_t)(nt * 16 + nlo) * H_DIM + khi * 8;
    const bf16* a_base0 = &hbuf[nlo][0];
    const bf16* a_base1 = &hbuf[16 + nlo][0];
#pragma unroll
    for (int kb = 0; kb < H_DIM; kb += 32) {
      ABFrag a0, a1, b;
      a0.q[0] = *(const uint4*)(a_base0 + kb + khi * 8);
      a0.q[1] = *(const uint4*)(a_base0 + kb + 16 + khi * 8);
      a1.q[0] = *(const uint4*)(a_base1 + kb + khi * 8);
      a1.q[1] = *(const uint4*)(a_base1 + kb + 16 + khi * 8);
      b.q[0]  = *(const uint4*)(b_base + kb);
      b.q[1]  = *(const uint4*)(b_base + kb + 16);
      c0 = wmma_bf16(a0, b, c0);
      c1 = wmma_bf16(a1, b, c1);
    }
#pragma unroll
    for (int r = 0; r < 8; ++r) {
      int m0 = khi * 8 + r;
      int m1 = 16 + khi * 8 + r;
      int ta = tokS[m0];
      int tb = tokS[m1];
      if (ta >= 0)
        atomicAdd(&out[(size_t)ta * D_DIM + nt * 16 + nlo], c0[r] * scS[m0]);
      if (tb >= 0)
        atomicAdd(&out[(size_t)tb * D_DIM + nt * 16 + nlo], c1[r] * scS[m1]);
    }
  }
}

extern "C" void kernel_launch(void* const* d_in, const int* in_sizes, int n_in,
                              void* d_out, int out_size, void* d_ws, size_t ws_size,
                              hipStream_t stream) {
  (void)n_in; (void)out_size; (void)ws_size;
  const float* x  = (const float*)d_in[0];
  const float* rw = (const float*)d_in[1];
  const float* rb = (const float*)d_in[2];
  const float* gw = (const float*)d_in[3];
  const float* uw = (const float*)d_in[4];
  const float* dw = (const float*)d_in[5];
  const float* sg = (const float*)d_in[6];
  const float* su = (const float*)d_in[7];
  const float* sd = (const float*)d_in[8];
  float* out = (float*)d_out;
  int N = in_sizes[0] / D_DIM;   // B*T tokens

  uintptr_t p = (uintptr_t)d_ws;
  auto alloc = [&](size_t bytes) -> void* {
    p = (p + 255) & ~(uintptr_t)255;
    void* r = (void*)p;
    p += bytes;
    return r;
  };
  bf16* xb   = (bf16*)alloc((size_t)N * D_DIM * sizeof(bf16));
  bf16* gwb  = (bf16*)alloc((size_t)E_NUM * H_DIM * D_DIM * sizeof(bf16));
  bf16* uwb  = (bf16*)alloc((size_t)E_NUM * H_DIM * D_DIM * sizeof(bf16));
  bf16* dwb  = (bf16*)alloc((size_t)E_NUM * D_DIM * H_DIM * sizeof(bf16));
  bf16* sgb  = (bf16*)alloc((size_t)HS_DIM * D_DIM * sizeof(bf16));
  bf16* sub  = (bf16*)alloc((size_t)HS_DIM * D_DIM * sizeof(bf16));
  bf16* sdb  = (bf16*)alloc((size_t)D_DIM * HS_DIM * sizeof(bf16));
  int*   tki    = (int*)alloc((size_t)N * K_TOP * sizeof(int));
  float* tks    = (float*)alloc((size_t)N * K_TOP * sizeof(float));
  int*   ptok   = (int*)alloc((size_t)N * K_TOP * sizeof(int));
  float* psc    = (float*)alloc((size_t)N * K_TOP * sizeof(float));
  int*   counts = (int*)alloc(E_NUM * sizeof(int));
  int*   offs   = (int*)alloc(E_NUM * sizeof(int));
  int*   cursor = (int*)alloc(E_NUM * sizeof(int));

  auto cvt = [&](const float* s, bf16* d, size_t n) {
    int n4 = (int)(n / 4);
    cvt_bf16_kernel<<<(n4 + 255) / 256, 256, 0, stream>>>(s, d, n4);
  };
  cvt(x,  xb,  (size_t)N * D_DIM);
  cvt(gw, gwb, (size_t)E_NUM * H_DIM * D_DIM);
  cvt(uw, uwb, (size_t)E_NUM * H_DIM * D_DIM);
  cvt(dw, dwb, (size_t)E_NUM * D_DIM * H_DIM);
  cvt(sg, sgb, (size_t)HS_DIM * D_DIM);
  cvt(su, sub, (size_t)HS_DIM * D_DIM);
  cvt(sd, sdb, (size_t)D_DIM * HS_DIM);

  zero_i32_kernel<<<1, 64, 0, stream>>>(counts, E_NUM);
  router_kernel<<<N, 256, 0, stream>>>(x, rw, rb, tki, tks, counts);
  offsets_kernel<<<1, 32, 0, stream>>>(counts, offs, cursor);
  scatter_kernel<<<(N * K_TOP + 255) / 256, 256, 0, stream>>>(tki, tks, offs, cursor,
                                                              ptok, psc, N * K_TOP);
  shared_expert_kernel<<<(N + M_TILE - 1) / M_TILE, 256, 0, stream>>>(xb, sgb, sub, sdb, out);
  dim3 eg((N + M_TILE - 1) / M_TILE, E_NUM);
  routed_expert_kernel<<<eg, 256, 0, stream>>>(xb, gwb, uwb, dwb, ptok, psc, offs, counts, out);
}
